// RNNContext_53910429500028
// MI455X (gfx1250) — compile-verified, matrix operand order checked by reading, not run on hardware
//
#include <hip/hip_runtime.h>

// Problem constants (from reference): B=1024, T=512, I=256, H=256
#define B_TOTAL 1024
#define T_LEN   512
#define I_DIM   256
#define H_DIM   256
#define G3      768         // 3*H
#define MB      16          // batch rows per workgroup (WMMA M)
#define NTHREADS 512        // 16 waves (wave32)
#define NWAVES  16
#define TILES_PER_WAVE 6    // 96 tiles / 16 waves

typedef __attribute__((ext_vector_type(16))) __bf16 v16bf;
typedef __attribute__((ext_vector_type(8)))  float  v8f;
typedef __attribute__((ext_vector_type(4)))  unsigned int u32x4;
typedef __attribute__((ext_vector_type(4)))  float  f32x4;

union BF16Frag {
    v16bf v;
    u32x4 q[2];
};

__device__ __forceinline__ unsigned short f32_to_bf16(float f) {
    unsigned int u = __float_as_uint(f);
    u += 0x7FFFu + ((u >> 16) & 1u);   // round-to-nearest-even
    return (unsigned short)(u >> 16);
}

__device__ __forceinline__ unsigned int pack_bf16x2(float a, float b) {
    return (unsigned int)f32_to_bf16(a) | ((unsigned int)f32_to_bf16(b) << 16);
}

__device__ __forceinline__ float sigmoidf_fast(float x) {
    return 1.0f / (1.0f + __expf(-x));
}

// ---------------------------------------------------------------------------
// Preprocess: f32 weights -> bf16 (row-major [768][256]) in workspace.
// ---------------------------------------------------------------------------
__global__ void convert_bf16_kernel(const float* __restrict__ src,
                                    unsigned short* __restrict__ dst, int n) {
    int i = blockIdx.x * blockDim.x + threadIdx.x;
    if (i < n) dst[i] = f32_to_bf16(src[i]);
}

// ---------------------------------------------------------------------------
// Persistent GRU: each workgroup owns MB=16 batch rows for all T steps.
// Two barriers per time step:
//   [GEMM] --barrier--> [gate math + stage x_{t+1}] --barrier--> next step
// ---------------------------------------------------------------------------
__global__ __launch_bounds__(NTHREADS)
void gru_persistent_kernel(const float* __restrict__ x,            // [B][T][I]
                           const int*   __restrict__ seq_len,      // [B]
                           const unsigned short* __restrict__ Wih, // [768][256] bf16
                           const unsigned short* __restrict__ Whh, // [768][256] bf16
                           const float* __restrict__ b_ih,         // [768]
                           const float* __restrict__ b_hh,         // [768]
                           float* __restrict__ out) {              // [B][H]
    __shared__ __align__(16) unsigned short A_x[MB * I_DIM];   // 8 KB  bf16 x_t
    __shared__ __align__(16) unsigned short A_h[MB * H_DIM];   // 8 KB  bf16 h
    __shared__ float XI[MB * G3];                              // 48 KB x proj
    __shared__ float GH[MB * G3];                              // 48 KB h proj

    const int tid    = threadIdx.x;
    const int lane   = tid & 31;
    const int wave   = tid >> 5;            // 0..15
    const int wgBase = blockIdx.x * MB;

    // Per-thread ownership for load/gate phases: row m_own, columns j0..j0+7
    const int m_own = wave;                 // 0..15
    const int j0    = (tid & 31) * 8;       // 0..248

    // WMMA lane geometry (16-bit operand layout, wave32)
    const int laneHalf = lane >> 4;         // 0: lanes 0-15, 1: lanes 16-31
    const int rowA     = lane & 15;
    const int kOffLane = laneHalf * 8;      // upper half-wave holds K+8
    const int mOffC    = laneHalf * 8;      // C/D: upper half-wave holds M+8

    // Recurrent state in registers (f32)
    float h_reg[8], hn_reg[8];
#pragma unroll
    for (int k = 0; k < 8; ++k) { h_reg[k] = 0.0f; hn_reg[k] = 0.0f; }

    // Zero the bf16 h operand buffer
    *(u32x4*)&A_h[m_own * H_DIM + j0] = (u32x4){0u, 0u, 0u, 0u};

    const int sl = seq_len[wgBase + m_own];

    // Preload biases for this thread's 8 columns (r/z/n segments)
    float bir[8], biz[8], bin_[8], bhr[8], bhz[8], bhn[8];
#pragma unroll
    for (int k = 0; k < 8; ++k) {
        int j = j0 + k;
        bir[k]  = b_ih[j];
        biz[k]  = b_ih[H_DIM + j];
        bin_[k] = b_ih[2 * H_DIM + j];
        bhr[k]  = b_hh[j];
        bhz[k]  = b_hh[H_DIM + j];
        bhn[k]  = b_hh[2 * H_DIM + j];
    }

    // Base pointer to this thread's x chunk (advance by I_DIM per step)
    const float* xp = x + ((size_t)(wgBase + m_own) * T_LEN) * I_DIM + j0;

    // Stage x_0 into LDS as bf16
    {
        f32x4 f0 = *(const f32x4*)(xp);
        f32x4 f1 = *(const f32x4*)(xp + 4);
        u32x4 pk;
        pk.x = pack_bf16x2(f0.x, f0.y);
        pk.y = pack_bf16x2(f0.z, f0.w);
        pk.z = pack_bf16x2(f1.x, f1.y);
        pk.w = pack_bf16x2(f1.z, f1.w);
        *(u32x4*)&A_x[m_own * I_DIM + j0] = pk;
        __builtin_prefetch((const void*)(xp + I_DIM), 0, 1);
    }
    __syncthreads();

#pragma unroll 1
    for (int t = 0; t < T_LEN; ++t) {
        // ---- Phase A: dual GEMM via WMMA ----
        // tiles 0..47 : XI = x_t @ W_ih^T   (A = A_x)
        // tiles 48..95: GH = h   @ W_hh^T   (A = A_h)
#pragma unroll 1
        for (int it = 0; it < TILES_PER_WAVE; ++it) {
            const int tile  = wave + NWAVES * it;        // 0..95
            const bool isH  = tile >= 48;
            const int ntile = isH ? (tile - 48) : tile;
            const unsigned short* __restrict__ W = isH ? Whh : Wih;
            const unsigned short* As = isH ? A_h : A_x;
            float* OUT = isH ? GH : XI;

            const int grow = ntile * 16 + rowA;          // gate (N) row in W
            v8f acc = (v8f){0.f, 0.f, 0.f, 0.f, 0.f, 0.f, 0.f, 0.f};
#pragma unroll
            for (int kb = 0; kb < 8; ++kb) {             // K = 256 = 8 * 32
                const int k0 = kb * 32 + kOffLane;
                BF16Frag a, b;
                a.q[0] = *(const u32x4*)&As[rowA * 256 + k0];
                a.q[1] = *(const u32x4*)&As[rowA * 256 + k0 + 16];
                b.q[0] = *(const u32x4*)&W[grow * 256 + k0];
                b.q[1] = *(const u32x4*)&W[grow * 256 + k0 + 16];
                acc = __builtin_amdgcn_wmma_f32_16x16x32_bf16(
                    false, a.v, false, b.v, (short)0, acc, false, false);
            }
            const int col = ntile * 16 + rowA;
#pragma unroll
            for (int v = 0; v < 8; ++v) {
                OUT[(v + mOffC) * G3 + col] = acc[v];
            }
        }
        __syncthreads();   // XI/GH ready; A_x/A_h reads complete

        // ---- Phase B: issue x_{t+1} loads, then gate math under them ----
        const bool haveNext = (t + 1) < T_LEN;
        f32x4 f0, f1;
        if (haveNext) {
            const float* xn = xp + (size_t)(t + 1) * I_DIM;
            f0 = *(const f32x4*)(xn);
            f1 = *(const f32x4*)(xn + 4);
            if ((t + 2) < T_LEN) {
                __builtin_prefetch((const void*)(xn + I_DIM), 0, 1);
            }
        }

        {
            const float* xiRow = &XI[m_own * G3];
            const float* ghRow = &GH[m_own * G3];
            unsigned int hb[4];
#pragma unroll
            for (int k = 0; k < 8; k += 2) {
                float hnew2[2];
#pragma unroll
                for (int s = 0; s < 2; ++s) {
                    const int kk = k + s;
                    const int j  = j0 + kk;
                    float r = sigmoidf_fast(xiRow[j] + bir[kk] +
                                            ghRow[j] + bhr[kk]);
                    float z = sigmoidf_fast(xiRow[H_DIM + j] + biz[kk] +
                                            ghRow[H_DIM + j] + bhz[kk]);
                    float hg = ghRow[2 * H_DIM + j] + bhn[kk];
                    float n  = tanhf(xiRow[2 * H_DIM + j] + bin_[kk] + r * hg);
                    float hnew = (1.0f - z) * n + z * h_reg[kk];
                    h_reg[kk] = hnew;
                    if (t < sl) hn_reg[kk] = hnew;
                    hnew2[s] = hnew;
                }
                hb[k >> 1] = pack_bf16x2(hnew2[0], hnew2[1]);
            }
            u32x4 pk = {hb[0], hb[1], hb[2], hb[3]};
            *(u32x4*)&A_h[m_own * H_DIM + j0] = pk;
        }

        if (haveNext) {
            u32x4 pk;
            pk.x = pack_bf16x2(f0.x, f0.y);
            pk.y = pack_bf16x2(f0.z, f0.w);
            pk.z = pack_bf16x2(f1.x, f1.y);
            pk.w = pack_bf16x2(f1.z, f1.w);
            *(u32x4*)&A_x[m_own * I_DIM + j0] = pk;
        }
        __syncthreads();   // A_h (and A_x) ready for next GEMM; XI/GH reads done
    }

    // ---- Write last-valid hidden state ----
#pragma unroll
    for (int k = 0; k < 8; ++k) {
        out[(size_t)(wgBase + m_own) * H_DIM + j0 + k] = hn_reg[k];
    }
}

// ---------------------------------------------------------------------------
extern "C" void kernel_launch(void* const* d_in, const int* in_sizes, int n_in,
                              void* d_out, int out_size, void* d_ws, size_t ws_size,
                              hipStream_t stream) {
    const float* x    = (const float*)d_in[0];
    const int*   sl   = (const int*)d_in[1];
    const float* Wih  = (const float*)d_in[2];
    const float* Whh  = (const float*)d_in[3];
    const float* bih  = (const float*)d_in[4];
    const float* bhh  = (const float*)d_in[5];

    unsigned short* WihB = (unsigned short*)d_ws;
    unsigned short* WhhB = WihB + (size_t)G3 * I_DIM;

    const int nW = G3 * I_DIM;  // 196608 per matrix
    convert_bf16_kernel<<<(nW + 255) / 256, 256, 0, stream>>>(Wih, WihB, nW);
    convert_bf16_kernel<<<(nW + 255) / 256, 256, 0, stream>>>(Whh, WhhB, nW);

    gru_persistent_kernel<<<B_TOTAL / MB, NTHREADS, 0, stream>>>(
        x, sl, WihB, WhhB, bih, bhh, (float*)d_out);
}
